// SpatialFrequencyAttentionBlock_10067403342448
// MI455X (gfx1250) — matrix operand order repaired and checked by reading, not compiled, not run on hardware
//
#include <hip/hip_runtime.h>

// ---------------------------------------------------------------------------
// SpatialFrequencyAttentionBlock forward for MI455X (gfx1250), wave32 + WMMA
// + Tensor Data Mover.
// GEMM-class ops (conv3x3 via 9-tap implicit GEMM, all 1x1 convs / linear
// projections) use v_wmma_f32_16x16x32_bf16 with f32 accumulation; activations
// are converted f32->bf16 (RNE) while staging into LDS. Pre-packed bf16 weight
// panels are staged global->LDS with TENSOR_LOAD_TO_LDS (TENSORcnt-tracked),
// overlapped with the activation conversion, with a VALU-copy fallback when
// the builtin is unavailable. Everything else is VALU.
//
// Input order assumption (JAX pytree = dicts flattened with sorted keys):
//  0:x | f1..f4 @1,13,25,37: a,mb,mbe,mg,mw,na,nb,ng,pb,pbe,pg,pw
//  fn: 49 b, 50 g | fu: 51 a, 52 b, 53 be, 54 g, 55 w
//  s1..s4 @56,61,66,71: a,b,be,g,w | sca: 76 w1, 77 w2 | ssa: 78 w
//  tca: 79 lb, 80 lg, 81 pb, 82 pw, 83 qkv
//  tsa: 84 lb, 85 lg, 86 pb, 87 pw, 88 qb, 89 qw
// Workspace requirement: ~540 MB.
// ---------------------------------------------------------------------------

#define B_      8
#define C_      64
#define H_      192
#define W_      192
#define PIX     36864          // H_*W_
#define T_ELEMS 18874368L      // B_*C_*PIX
#define NPATCH  4608           // B_*24*24
#define EPS_    1e-5f
#define APITCH  136            // LDS row pitch (bf16) for the A tile

typedef __attribute__((ext_vector_type(16))) __bf16 v16bf;
typedef __attribute__((ext_vector_type(8)))  float  v8f;

struct __align__(16) U4 { unsigned a, b, c, d; };
union Frag { v16bf v; U4 q[2]; };

__device__ __forceinline__ unsigned short f2bf(float f) {   // f32 -> bf16 RNE
  unsigned u = __builtin_bit_cast(unsigned, f);
  return (unsigned short)((u + 0x7fffu + ((u >> 16) & 1u)) >> 16);
}
__device__ __forceinline__ float sigm(float x) { return 1.f / (1.f + __expf(-x)); }
__device__ __forceinline__ v8f vzero() { v8f z; for (int i = 0; i < 8; ++i) z[i] = 0.f; return z; }

// ----------------------- Tensor Data Mover staging -------------------------
// Issue a TDM global->LDS copy of `rows` x `k` bf16 elements (dense in global,
// optional LDS row padding via the D# pad fields), then later wait on
// TENSORcnt. D# layout per CDNA5 ISA 8.3/8.4 (groups 2/3 zero => <=2D tensor).

#if __has_builtin(__builtin_amdgcn_tensor_load_to_lds) && \
    __has_builtin(__builtin_amdgcn_s_wait_tensorcnt)
#define USE_TDM 1
typedef unsigned int u32x4 __attribute__((ext_vector_type(4)));
typedef int          i32x4 __attribute__((ext_vector_type(4)));
typedef int          i32x8 __attribute__((ext_vector_type(8)));

__device__ __forceinline__ void tdm_issue_2d(unsigned short* ldsDst,
    const unsigned short* gsrc, unsigned k, unsigned rows, unsigned cfg0) {
  u32x4 g0; i32x8 g1; i32x4 z4;
  z4[0] = z4[1] = z4[2] = z4[3] = 0;
  unsigned long long ga = (unsigned long long)gsrc;
  unsigned long long s1 = (unsigned long long)k * rows;     // dim1 stride
  g0[0] = 1u;                                               // count=1, user D#
  g0[1] = (unsigned)(unsigned long long)ldsDst;             // lds_addr (bytes)
  g0[2] = (unsigned)ga;                                     // global_addr[31:0]
  g0[3] = ((unsigned)(ga >> 32) & 0x01ffffffu) | (2u << 30); // addr[56:32]|type=2
  g1[0] = (int)cfg0;                            // data_size(=2B) | pad config
  g1[1] = (int)((k & 0xffffu) << 16);           // tensor_dim0[15:0]
  g1[2] = (int)((k >> 16) | ((rows & 0xffffu) << 16)); // dim0 hi | tensor_dim1 lo
  g1[3] = (int)(((rows >> 16) & 0xffffu) | ((k & 0xffffu) << 16)); // dim1 hi | tile_dim0
  g1[4] = (int)(rows & 0xffffu);                // tile_dim1 (tile_dim2 = 0)
  g1[5] = (int)(unsigned)k;                     // tensor_dim0_stride[31:0]
  g1[6] = (int)((unsigned)(s1 & 0xffffu) << 16); // dim0 stride hi(0) | dim1 stride lo
  g1[7] = (int)(unsigned)(s1 >> 16);            // tensor_dim1_stride[47:16]
#if defined(__clang_major__) && __clang_major__ >= 23
  i32x8 z8;
  for (int i = 0; i < 8; ++i) z8[i] = 0;
  __builtin_amdgcn_tensor_load_to_lds(g0, g1, z4, z4, z8, 0);
#else
  __builtin_amdgcn_tensor_load_to_lds(g0, g1, z4, z4, 0);
#endif
}
__device__ __forceinline__ void tdm_wait() {
  __builtin_amdgcn_s_wait_tensorcnt((short)0);
}
#define TDM_CFG_DENSE   (1u << 16)                 // data_size = 2 bytes
// pad: enable | interval (2<<code dwords between pads) | amount (code+1 dwords)
#define TDM_CFG_PAD(ivalCode) ((1u << 16) | (1u << 20) | ((unsigned)(ivalCode) << 22) | (3u << 25))
#endif

// ------------------------------- utilities ---------------------------------

__global__ __launch_bounds__(256) void k_zero(float* p, long n) {
  long i = (long)blockIdx.x * 256 + threadIdx.x;
  if (i < n) p[i] = 0.f;
}

__global__ __launch_bounds__(256) void k_pack_slice(const float* __restrict__ w,
    unsigned short* __restrict__ o, int srcK, int Kc, int off, int n) {
  int i = blockIdx.x * 256 + threadIdx.x;
  if (i >= n) return;
  int co = i / Kc, ci = i - co * Kc;
  o[i] = f2bf(w[(size_t)co * srcK + off + ci]);
}

// (64,64,3,3) -> bf16 [tap][co][ci]
__global__ __launch_bounds__(256) void k_pack3x3(const float* __restrict__ w,
    unsigned short* __restrict__ o) {
  int i = blockIdx.x * 256 + threadIdx.x;
  if (i >= 9 * 64 * 64) return;
  int tap = i / 4096, r = i - tap * 4096;
  int co = r >> 6, ci = r & 63;
  o[i] = f2bf(w[((size_t)co * 64 + ci) * 9 + tap]);
}

// ------------------------- WMMA conv3x3 (stage) ----------------------------
// grid (W/64, H, B), block 128 (4 waves); wave owns 16 pixels x 64 out-ch.
__global__ __launch_bounds__(128) void k_conv3x3(const float* __restrict__ x,
    const unsigned short* __restrict__ wp, const float* __restrict__ bias,
    float* __restrict__ out) {
  __shared__ __align__(16) unsigned short Ain[3][68][64]; // [row][col(halo)][ci]
  __shared__ __align__(16) unsigned short Wt[9][64][64];  // [tap][co][ci]
  const int t = threadIdx.x, lane = t & 31, wv = t >> 5;
  const int wbase = blockIdx.x * 64, h = blockIdx.y, b = blockIdx.z;

#ifdef USE_TDM
  // DMA the 73.7KB bf16 weight panel into LDS; overlap with halo conversion.
  if (t < 32) tdm_issue_2d(&Wt[0][0][0], wp, 9 * 64 * 64, 1, TDM_CFG_DENSE);
#endif
  { // halo tile, f32 -> bf16, zero-pad borders (VALU work overlapping the TDM)
    for (int i = t; i < 3 * 66 * 64; i += 128) {
      int ci = i / (3 * 66), rc = i - ci * 3 * 66;
      int r = rc / 66, c = rc - r * 66;
      int hh = h + r - 1, ww = wbase + c - 1;
      float v = 0.f;
      if (hh >= 0 && hh < H_ && (unsigned)ww < (unsigned)W_)
        v = x[(((size_t)b * C_ + ci) * H_ + hh) * W_ + ww];
      Ain[r][c][ci] = f2bf(v);
    }
  }
#ifdef USE_TDM
  if (t < 32) tdm_wait();
#else
  { // fallback: VALU copy of the pre-packed bf16 weights, 16B chunks
    const U4* s = (const U4*)wp;
    U4* d = (U4*)&Wt[0][0][0];
    for (int i = t; i < 9 * 64 * 64 / 8; i += 128) d[i] = s[i];
  }
#endif
  __syncthreads();

  const int mrow = lane & 15, khalf = lane >> 4;
  v8f acc[4];
  for (int nb = 0; nb < 4; ++nb) acc[nb] = vzero();

  for (int tap = 0; tap < 9; ++tap) {
    const int ky = tap / 3, kx = tap - ky * 3;
    for (int kc = 0; kc < 2; ++kc) {
      Frag af;  // A: lane row m, K subsets {kh*8..+7} and {+16..}
      const unsigned short* ap = &Ain[ky][wv * 16 + mrow + kx][kc * 32 + khalf * 8];
      af.q[0] = *(const U4*)ap;
      af.q[1] = *(const U4*)(ap + 16);
      for (int nb = 0; nb < 4; ++nb) {
        Frag bf;  // B: lane col co, K subsets {kh*8..+7} and {+16..}
        const unsigned short* bp = &Wt[tap][nb * 16 + mrow][kc * 32 + khalf * 8];
        bf.q[0] = *(const U4*)bp;
        bf.q[1] = *(const U4*)(bp + 16);
        acc[nb] = __builtin_amdgcn_wmma_f32_16x16x32_bf16(
            false, af.v, false, bf.v, (short)0, acc[nb], false, false);
      }
    }
  }
  const int p0 = wbase + wv * 16 + khalf * 8;
  for (int nb = 0; nb < 4; ++nb) {
    int co = nb * 16 + mrow;
    float bv = bias[co];
    float* o = &out[(((size_t)b * C_ + co) * H_ + h) * W_ + p0];
    *(float4*)o       = make_float4(acc[nb][0] + bv, acc[nb][1] + bv, acc[nb][2] + bv, acc[nb][3] + bv);
    *(float4*)(o + 4) = make_float4(acc[nb][4] + bv, acc[nb][5] + bv, acc[nb][6] + bv, acc[nb][7] + bv);
  }
}

// --------------------- generic WMMA GEMM (1x1 conv / linear) ---------------
// out[m, co] = sum_ci A[m, ci] * W[co, ci] (+bias); M tile 64 (wave = 16).
// inTok:  0 -> channel-major x[(b*K + ci)*pixPB + p] (concat via x1/K0, roll remap)
//         1 -> token-major   x[m*K + ci]
// outTok: 0 -> out[(b*Ntot + co)*pixPB + p] ; 1 -> out[m*Ntot + co]
// Weight panel LDS pitch is K+8 bf16; on the TDM path the hardware inserts the
// 4-DWORD bank-skew pad per row (pad_interval = K/2 dwords, pad_amount = 4).
__global__ __launch_bounds__(128) void k_gemm1x1(const float* __restrict__ x0,
    const float* __restrict__ x1, int K0, int K,
    const unsigned short* __restrict__ wp, const float* __restrict__ bias,
    float* __restrict__ out, int pixPB, int Ntot,
    int inTok, int outTok, int remap, int accum) {
  __shared__ __align__(16) unsigned short As[64 * APITCH];
  __shared__ __align__(16) unsigned short Ws[192 * APITCH];
  const int t = threadIdx.x, lane = t & 31, wv = t >> 5;
  const long mbase = (long)blockIdx.x * 64;
  const int wpitch = K + 8;
  __builtin_prefetch(wp, 0, 0);   // global_prefetch_b8 for the weight pack

#ifdef USE_TDM
  if (t < 32)
    tdm_issue_2d(Ws, wp, (unsigned)K, (unsigned)Ntot, TDM_CFG_PAD(K == 64 ? 4 : 5));
#endif
  for (int i = t; i < 64 * K; i += 128) {   // A tile: load f32, convert, stage
    int ci = i >> 6, mm = i & 63;
    long m = mbase + mm;
    float v;
    if (inTok) {
      v = x0[(size_t)m * K + ci];
    } else {
      long bb = m / pixPB;
      int p = (int)(m - bb * pixPB);
      if (remap) { int r = p >> 3, c = p & 7; p = (((r + 7) & 7) << 3) | c; } // roll(h,1)
      v = (ci < K0) ? x0[((size_t)bb * K0 + ci) * pixPB + p]
                    : x1[((size_t)bb * (K - K0) + (ci - K0)) * pixPB + p];
    }
    As[mm * APITCH + ci] = f2bf(v);
  }
#ifdef USE_TDM
  if (t < 32) tdm_wait();
#else
  for (int i = t; i < Ntot * K; i += 128) {
    int co = i / K, ci = i - co * K;
    Ws[co * wpitch + ci] = wp[i];
  }
#endif
  __syncthreads();

  const int mrow = lane & 15, khalf = lane >> 4;
  const int kit = K >> 5, nt = Ntot >> 4;
  for (int nb = 0; nb < nt; ++nb) {
    v8f acc = vzero();
    for (int kc = 0; kc < kit; ++kc) {
      Frag af, bf;
      const unsigned short* ap = &As[(wv * 16 + mrow) * APITCH + kc * 32 + khalf * 8];
      af.q[0] = *(const U4*)ap;
      af.q[1] = *(const U4*)(ap + 16);
      const unsigned short* bp = &Ws[(nb * 16 + mrow) * wpitch + kc * 32 + khalf * 8];
      bf.q[0] = *(const U4*)bp;
      bf.q[1] = *(const U4*)(bp + 16);
      acc = __builtin_amdgcn_wmma_f32_16x16x32_bf16(
          false, af.v, false, bf.v, (short)0, acc, false, false);
    }
    int co = nb * 16 + mrow;
    float bv = bias ? bias[co] : 0.f;
    for (int r = 0; r < 8; ++r) {
      long m = mbase + wv * 16 + khalf * 8 + r;
      size_t addr;
      if (outTok) addr = (size_t)m * Ntot + co;
      else {
        long bb = m / pixPB;
        int p = (int)(m - bb * pixPB);
        addr = ((size_t)bb * Ntot + co) * pixPB + p;
      }
      float val = acc[r] + bv;
      if (accum) val += out[addr];
      out[addr] = val;
    }
  }
}

// ------------------------------ batchnorm ----------------------------------

__global__ __launch_bounds__(256) void k_bn_stats(const float* __restrict__ x,
    int P, long cnt, float* __restrict__ stats) {
  int ch = blockIdx.x;
  __shared__ float r1[256], r2[256];
  float s = 0.f, s2 = 0.f;
  for (int it = 0; it < 16; ++it) {
    long j = (long)blockIdx.y * 4096 + it * 256 + threadIdx.x;
    if (j < cnt) {
      long bb = j / P;
      int p = (int)(j - bb * P);
      float v = x[((size_t)bb * C_ + ch) * P + p];
      s += v; s2 += v * v;
    }
  }
  r1[threadIdx.x] = s; r2[threadIdx.x] = s2;
  __syncthreads();
  for (int o = 128; o > 0; o >>= 1) {
    if (threadIdx.x < o) { r1[threadIdx.x] += r1[threadIdx.x + o]; r2[threadIdx.x] += r2[threadIdx.x + o]; }
    __syncthreads();
  }
  if (threadIdx.x == 0) { atomicAdd(&stats[ch], r1[0]); atomicAdd(&stats[C_ + ch], r2[0]); }
}

__global__ __launch_bounds__(256) void k_bn_act(const float* __restrict__ x,
    const float* __restrict__ stats, const float* __restrict__ g,
    const float* __restrict__ be, const float* __restrict__ a,
    float invc, float* __restrict__ out, int P, long total) {
  long i = (long)blockIdx.x * 256 + threadIdx.x;
  if (i >= total) return;
  int ch = (int)((i / P) % C_);
  float m = stats[ch] * invc, var = stats[C_ + ch] * invc - m * m;
  float y = g[ch] * (x[i] - m) * rsqrtf(var + EPS_) + be[ch];
  if (a) { float al = *a; y = y >= 0.f ? y : al * y; }
  out[i] = y;
}

__global__ __launch_bounds__(256) void k_bn_add(const float* __restrict__ x,
    const float* __restrict__ stats, const float* __restrict__ g,
    const float* __restrict__ be, const float* __restrict__ idn,
    float invc, float* __restrict__ out) {
  long i = (long)blockIdx.x * 256 + threadIdx.x;
  if (i >= T_ELEMS) return;
  int ch = (int)((i / PIX) % C_);
  float m = stats[ch] * invc, var = stats[C_ + ch] * invc - m * m;
  out[i] = g[ch] * (x[i] - m) * rsqrtf(var + EPS_) + be[ch] + idn[i];
}

// ------------------------- broadcast multiplies ----------------------------

__global__ __launch_bounds__(256) void k_mul_bc(float* __restrict__ x,
    const float* __restrict__ map, int mode) {
  long i = (long)blockIdx.x * 256 + threadIdx.x;
  if (i >= T_ELEMS) return;
  long b = i / (64L * PIX);
  if (mode == 0) { int ch = (int)((i / PIX) % 64); x[i] *= map[b * 64 + ch]; }
  else           { int p  = (int)(i % PIX);        x[i] *= map[b * PIX + p]; }
}

// --------------------------------- sca -------------------------------------

__global__ __launch_bounds__(256) void k_pool_bc(const float* __restrict__ x,
    float* __restrict__ pm, float* __restrict__ px) {
  int ch = blockIdx.x, b = blockIdx.y;
  const float* p = x + ((size_t)b * 64 + ch) * PIX;
  float s = 0.f, m = -1e30f;
  for (int i = threadIdx.x; i < PIX; i += 256) { float v = p[i]; s += v; m = fmaxf(m, v); }
  __shared__ float r1[256], r2[256];
  r1[threadIdx.x] = s; r2[threadIdx.x] = m;
  __syncthreads();
  for (int o = 128; o > 0; o >>= 1) {
    if (threadIdx.x < o) { r1[threadIdx.x] += r1[threadIdx.x + o]; r2[threadIdx.x] = fmaxf(r2[threadIdx.x], r2[threadIdx.x + o]); }
    __syncthreads();
  }
  if (threadIdx.x == 0) { pm[b * 64 + ch] = r1[0] * (1.f / PIX); px[b * 64 + ch] = r2[0]; }
}

__global__ __launch_bounds__(64) void k_sca(const float* __restrict__ mean,
    const float* __restrict__ mx, const float* __restrict__ w1,
    const float* __restrict__ w2, float* __restrict__ sig) {
  int b = blockIdx.x, c = threadIdx.x;
  __shared__ float ym[64], yx[64], hm[4], hx[4];
  ym[c] = mean[b * 64 + c]; yx[c] = mx[b * 64 + c];
  __syncthreads();
  if (c < 4) {
    float sm = 0.f, sx = 0.f;
    for (int i = 0; i < 64; ++i) { sm += w1[c * 64 + i] * ym[i]; sx += w1[c * 64 + i] * yx[i]; }
    hm[c] = fmaxf(sm, 0.f); hx[c] = fmaxf(sx, 0.f);
  }
  __syncthreads();
  float o = 0.f;
  for (int j = 0; j < 4; ++j) o += w2[c * 4 + j] * (hm[j] + hx[j]);
  sig[b * 64 + c] = sigm(o);
}

// --------------------------------- tca -------------------------------------

__global__ __launch_bounds__(256) void k_ln_ch(const float* __restrict__ x,
    const float* __restrict__ lg, const float* __restrict__ lb,
    float* __restrict__ out) {
  long i = (long)blockIdx.x * 256 + threadIdx.x;
  if (i >= 8L * PIX) return;
  long b = i / PIX; int p = (int)(i % PIX);
  const float* xp = x + (size_t)b * 64 * PIX + p;
  float v[64], s = 0.f;
  for (int c = 0; c < 64; ++c) { v[c] = xp[(size_t)c * PIX]; s += v[c]; }
  float m = s * (1.f / 64), vs = 0.f;
  for (int c = 0; c < 64; ++c) { float d = v[c] - m; vs += d * d; }
  float inv = rsqrtf(vs * (1.f / 64) + EPS_);
  float* op = out + (size_t)b * 64 * PIX + p;
  for (int c = 0; c < 64; ++c) op[(size_t)c * PIX] = lg[c] * (v[c] - m) * inv + lb[c];
}

// grid (B*8 heads, 72 slabs), 64 threads; G[bh][d][e] += q.k over 512 pixels
__global__ __launch_bounds__(64) void k_tca_gram(const float* __restrict__ qkv,
    float* __restrict__ G, float* __restrict__ mv) {
  int bh = blockIdx.x, b = bh >> 3, h = bh & 7;
  int n0 = blockIdx.y * 512, t = threadIdx.x, d = t >> 3, e = t & 7;
  const float* q = qkv + ((size_t)(b * 192 + h * 8 + d)) * PIX + n0;
  const float* k = qkv + ((size_t)(b * 192 + 64 + h * 8 + e)) * PIX + n0;
  float s = 0.f;
  for (int i = 0; i < 512; ++i) s += q[i] * k[i];
  atomicAdd(&G[(size_t)bh * 64 + t], s);
  if (t < 8) {
    const float* v = qkv + ((size_t)(b * 192 + 128 + h * 8 + t)) * PIX + n0;
    float sv = 0.f;
    for (int i = 0; i < 512; ++i) sv += v[i];
    atomicAdd(&mv[b * 64 + h * 8 + t], sv * (1.f / PIX));
  }
}

__global__ __launch_bounds__(64) void k_tca_finish(const float* __restrict__ G,
    const float* __restrict__ mv, const float* __restrict__ pw,
    const float* __restrict__ pb, float* __restrict__ sig) {
  int b = blockIdx.x, t = threadIdx.x, h = t >> 3, d = t & 7;
  __shared__ float am[64];
  const float* row = &G[(((size_t)b * 8 + h) * 8 + d) * 8];
  const float sc = 0.35355339059f;
  float mx = -1e30f;
  for (int e = 0; e < 8; ++e) mx = fmaxf(mx, row[e] * sc);
  float den = 0.f, num = 0.f;
  for (int e = 0; e < 8; ++e) {
    float w = __expf(row[e] * sc - mx);
    den += w; num += w * mv[b * 64 + h * 8 + e];
  }
  am[h * 8 + d] = num / den;
  __syncthreads();
  float z = pb[t];
  for (int c = 0; c < 64; ++c) z += pw[t * 64 + c] * am[c];
  sig[b * 64 + t] = sigm(z);
}

// --------------------------------- ssa -------------------------------------

__global__ __launch_bounds__(256) void k_chanpool(const float* __restrict__ x,
    float* __restrict__ y) {
  long i = (long)blockIdx.x * 256 + threadIdx.x;
  if (i >= 8L * PIX) return;
  long b = i / PIX; int p = (int)(i % PIX);
  const float* xp = x + (size_t)b * 64 * PIX + p;
  float s = 0.f, m = -1e30f;
  for (int c = 0; c < 64; ++c) { float v = xp[(size_t)c * PIX]; s += v; m = fmaxf(m, v); }
  y[((size_t)b * 2 + 0) * PIX + p] = s * (1.f / 64);
  y[((size_t)b * 2 + 1) * PIX + p] = m;
}

__global__ __launch_bounds__(256) void k_conv7(const float* __restrict__ y,
    const float* __restrict__ w, float* __restrict__ map) {
  long i = (long)blockIdx.x * 256 + threadIdx.x;
  if (i >= 8L * PIX) return;
  int b = (int)(i / PIX), p = (int)(i % PIX);
  int h = p / 192, x0 = p % 192;
  float s = 0.f;
  for (int ic = 0; ic < 2; ++ic)
    for (int ky = 0; ky < 7; ++ky) {
      int hh = h + ky - 3;
      if (hh < 0 || hh >= 192) continue;
      for (int kx = 0; kx < 7; ++kx) {
        int ww = x0 + kx - 3;
        if (ww < 0 || ww >= 192) continue;
        s += y[((size_t)b * 2 + ic) * PIX + hh * 192 + ww] * w[(ic * 7 + ky) * 7 + kx];
      }
    }
  map[i] = sigm(s);
}

// --------------------------------- tsa -------------------------------------

__global__ __launch_bounds__(256) void k_ln_win(const float* __restrict__ x,
    const float* __restrict__ lg, const float* __restrict__ lb,
    float* __restrict__ out) {
  long i = (long)blockIdx.x * 256 + threadIdx.x;
  if (i >= 294912L) return;
  long wid = i >> 6; int t = (int)(i & 63);
  int b = (int)(wid / 576), w2 = (int)(wid % 576), wh = w2 / 24, ww = w2 % 24;
  int hp = wh * 8 + (t >> 3), wp = ww * 8 + (t & 7);
  const float* xp = x + (size_t)b * 64 * PIX + hp * 192 + wp;
  float v[64], s = 0.f;
  for (int c = 0; c < 64; ++c) { v[c] = xp[(size_t)c * PIX]; s += v[c]; }
  float m = s * (1.f / 64), vs = 0.f;
  for (int c = 0; c < 64; ++c) { float d = v[c] - m; vs += d * d; }
  float inv = rsqrtf(vs * (1.f / 64) + EPS_);
  float* op = out + (size_t)i * 64;
  for (int c = 0; c < 64; ++c) op[c] = lg[c] * (v[c] - m) * inv + lb[c];
}

__global__ __launch_bounds__(64) void k_rowmean(const float* __restrict__ pw,
    const float* __restrict__ pb, float* __restrict__ u) {
  int c = threadIdx.x;
  float s = 0.f;
  for (int co = 0; co < 64; ++co) s += pw[co * 64 + c];
  u[c] = s * (1.f / 64);
  if (c == 0) {
    float t = 0.f;
    for (int j = 0; j < 64; ++j) t += pb[j];
    u[64] = t * (1.f / 64);
  }
}

// block = one 8x8 window (64 tokens x 192 qkv staged in LDS)
__global__ __launch_bounds__(256) void k_win_attn(const float* __restrict__ qkv,
    const float* __restrict__ u, float* __restrict__ map) {
  __shared__ float qs[64][192];
  __shared__ float accs[64];
  int wid = blockIdx.x, t = threadIdx.x;
  size_t tokbase = (size_t)wid * 64;
  for (int i = t; i < 64 * 192; i += 256) qs[i / 192][i % 192] = qkv[tokbase * 192 + i];
  if (t < 64) accs[t] = 0.f;
  __syncthreads();
  const float sc = 0.35355339059f;
  for (int pr = t; pr < 512; pr += 256) {
    int h = pr >> 6, tt = pr & 63;
    const float* qr = &qs[tt][h * 8];
    float mx = -1e30f;
    for (int uu = 0; uu < 64; ++uu) {
      const float* kr = &qs[uu][64 + h * 8];
      float s = 0.f;
      for (int d = 0; d < 8; ++d) s += qr[d] * kr[d];
      mx = fmaxf(mx, s * sc);
    }
    float den = 0.f;
    for (int uu = 0; uu < 64; ++uu) {
      const float* kr = &qs[uu][64 + h * 8];
      float s = 0.f;
      for (int d = 0; d < 8; ++d) s += qr[d] * kr[d];
      den += __expf(s * sc - mx);
    }
    float o[8];
    for (int d = 0; d < 8; ++d) o[d] = 0.f;
    for (int uu = 0; uu < 64; ++uu) {
      const float* kr = &qs[uu][64 + h * 8];
      float s = 0.f;
      for (int d = 0; d < 8; ++d) s += qr[d] * kr[d];
      float wgt = __expf(s * sc - mx);
      const float* vr = &qs[uu][128 + h * 8];
      for (int d = 0; d < 8; ++d) o[d] += wgt * vr[d];
    }
    float part = 0.f;
    for (int d = 0; d < 8; ++d) part += u[h * 8 + d] * o[d];
    atomicAdd(&accs[tt], part / den);
  }
  __syncthreads();
  if (t < 64) {
    int b = wid / 576, w2 = wid % 576, wh = w2 / 24, ww = w2 % 24;
    int hp = wh * 8 + (t >> 3), wp = ww * 8 + (t & 7);
    map[(size_t)b * PIX + hp * 192 + wp] = sigm(accs[t] + u[64]);
  }
}

// --------------------------------- fdp -------------------------------------

__global__ __launch_bounds__(256) void k_fdp_split(const float* __restrict__ x,
    float* __restrict__ xl, float* __restrict__ xh) {
  int i = blockIdx.x * 256 + threadIdx.x;
  if (i >= NPATCH * 64) return;
  int pid = i % NPATCH, ch = i / NPATCH;
  int b = pid / 576, pr = (pid % 576) / 24, pc = pid % 24;
  const float* xp = x + (((size_t)b * 64 + ch) * 192 + pr * 8) * 192 + pc * 8;
  float v[64];
  for (int r = 0; r < 8; ++r)
    for (int c = 0; c < 8; ++c) v[r * 8 + c] = xp[r * 192 + c];
  float* xlp = xl + (size_t)pid * 1024 + ch * 16;
  float* xhp = xh + (size_t)pid * 4096 + ch * 64;
  for (int qr = 0; qr < 4; ++qr)
    for (int qc = 0; qc < 4; ++qc) {
      int r0 = 2 * qr, c0 = 2 * qc;
      float m = 0.25f * (v[r0 * 8 + c0] + v[r0 * 8 + c0 + 1] + v[(r0 + 1) * 8 + c0] + v[(r0 + 1) * 8 + c0 + 1]);
      xlp[qr * 4 + qc] = m;
      xhp[r0 * 8 + c0] = v[r0 * 8 + c0] - m;
      xhp[r0 * 8 + c0 + 1] = v[r0 * 8 + c0 + 1] - m;
      xhp[(r0 + 1) * 8 + c0] = v[(r0 + 1) * 8 + c0] - m;
      xhp[(r0 + 1) * 8 + c0 + 1] = v[(r0 + 1) * 8 + c0 + 1] - m;
    }
}

__global__ __launch_bounds__(256) void k_fdp_merge(const float* __restrict__ xl,
    const float* __restrict__ xh, float* __restrict__ out) {
  long i = (long)blockIdx.x * 256 + threadIdx.x;
  if (i >= T_ELEMS) return;
  int w = (int)(i % 192); long r0 = i / 192;
  int h = (int)(r0 % 192); long r1 = r0 / 192;
  int ch = (int)(r1 % 64), b = (int)(r1 / 64);
  int pid = (b * 24 + h / 8) * 24 + w / 8;
  int r = h & 7, c = w & 7;
  out[i] = xl[(size_t)pid * 1024 + ch * 16 + (r >> 1) * 4 + (c >> 1)] +
           xh[(size_t)pid * 4096 + ch * 64 + r * 8 + c];
}

// ------------------------------- host side ---------------------------------

static inline dim3 NB(long n, int b) { return dim3((unsigned)((n + b - 1) / b)); }

extern "C" void kernel_launch(void* const* d_in, const int* in_sizes, int n_in,
                              void* d_out, int out_size, void* d_ws, size_t ws_size,
                              hipStream_t stream) {
  (void)in_sizes; (void)n_in; (void)out_size; (void)ws_size;
  const float* X = (const float*)d_in[0];
  auto P_ = [&](int i) { return (const float*)d_in[i]; };

  float* ws   = (float*)d_ws;
  float* BUFS = ws;
  float* BUFF = ws + T_ELEMS;
  float* T0   = ws + 2 * T_ELEMS;
  float* T1   = ws + 3 * T_ELEMS;
  float* QKV  = ws + 4 * T_ELEMS;            // 3*T; aliased with fdp scratch below
  float* XL   = QKV;                         // 4608*64*16
  float* XL2  = QKV + 4718592L;
  float* XH   = QKV + 2 * 4718592L;          // 4608*64*64
  float* XH2  = XH + T_ELEMS;
  float* SM    = ws + 7 * T_ELEMS;
  float* stats = SM;                         // 128
  float* G     = SM + 256;                   // 4096
  float* MV    = SM + 4352;                  // 512 (zeroed together with G)
  float* SIGA  = SM + 4864;
  float* SIGB  = SM + 5376;
  float* PMEAN = SM + 5888;
  float* PMAX  = SM + 6400;
  float* UVEC  = SM + 6912;                  // 65 used
  float* MAPA  = SM + 7040;                  // 8*PIX
  float* MAPB  = MAPA + 294912L;
  float* Y2    = MAPB + 294912L;             // 8*2*PIX
  unsigned short* WPK = (unsigned short*)(Y2 + 589824L);
  unsigned short* W3P[4], *MWP[4], *PWA[4], *PWB[4];
  for (int i = 0; i < 4; ++i) {
    W3P[i] = WPK + (size_t)i * 36864;
    MWP[i] = WPK + 147456 + (size_t)i * 4096;
    PWA[i] = WPK + 163840 + (size_t)i * 4096;
    PWB[i] = WPK + 180224 + (size_t)i * 4096;
  }
  unsigned short* QKVP = WPK + 196608;
  unsigned short* FUP  = WPK + 208896;
  unsigned short* QWP  = WPK + 217088;

  // ---- weight packing (f32 -> bf16, GEMM-friendly layouts) ----
  for (int s = 0; s < 4; ++s)
    k_pack3x3<<<NB(36864, 256), 256, 0, stream>>>(P_(56 + s * 5 + 4), W3P[s]);
  for (int f = 0; f < 4; ++f) {
    int fb = 1 + f * 12;
    k_pack_slice<<<NB(4096, 256), 256, 0, stream>>>(P_(fb + 4),  MWP[f], 64, 64, 0, 4096);
    k_pack_slice<<<NB(4096, 256), 256, 0, stream>>>(P_(fb + 11), PWA[f], 128, 64, 0, 4096);
    k_pack_slice<<<NB(4096, 256), 256, 0, stream>>>(P_(fb + 11), PWB[f], 128, 64, 64, 4096);
  }
  k_pack_slice<<<NB(12288, 256), 256, 0, stream>>>(P_(83), QKVP, 64, 64, 0, 12288);
  k_pack_slice<<<NB(8192, 256), 256, 0, stream>>>(P_(55), FUP, 128, 128, 0, 8192);
  k_pack_slice<<<NB(12288, 256), 256, 0, stream>>>(P_(89), QWP, 64, 64, 0, 12288);

  auto gemm = [&](const float* x0, const float* x1, int K0, int K,
                  const unsigned short* wp, const float* bias, float* out,
                  long Mtot, int pixPB, int Ntot, int inTok, int outTok,
                  int remap, int accum) {
    k_gemm1x1<<<NB(Mtot, 64), 128, 0, stream>>>(x0, x1, K0, K, wp, bias, out,
                                                pixPB, Ntot, inTok, outTok, remap, accum);
  };
  auto bnorm = [&](const float* src, float* dst, const float* g, const float* be,
                   const float* a, long nB, int Pp) {
    long cnt = nB * Pp;
    k_zero<<<1, 256, 0, stream>>>(stats, 128);
    k_bn_stats<<<dim3(64, (unsigned)((cnt + 4095) / 4096)), 256, 0, stream>>>(src, Pp, cnt, stats);
    k_bn_act<<<NB(cnt * 64, 256), 256, 0, stream>>>(src, stats, g, be, a,
                                                    1.f / (float)cnt, dst, Pp, cnt * 64);
  };
  auto stage = [&](const float* inx, float* outs, int si) {
    int sb = 56 + si * 5;
    k_conv3x3<<<dim3(3, 192, 8), 128, 0, stream>>>(inx, W3P[si], P_(sb + 1), T0);
    bnorm(T0, outs, P_(sb + 3), P_(sb + 2), P_(sb + 0), 8, PIX);
  };
  auto fstage = [&](const float* inx, float* outf, int fi) {
    int fb = 1 + fi * 12;
    k_fdp_split<<<NB(NPATCH * 64L, 256), 256, 0, stream>>>(inx, XL, XH);
    gemm(XL, XL, 64, 64, MWP[fi], P_(fb + 1), XL2, 73728, 16, 64, 0, 0, 0, 0);
    bnorm(XL2, XL2, P_(fb + 3), P_(fb + 2), P_(fb + 0), 4608, 16);
    gemm(XH, XH, 64, 64, PWA[fi], P_(fb + 8), XH2, 294912, 64, 64, 0, 0, 0, 0);
    gemm(XH, XH, 64, 64, PWB[fi], nullptr,   XH2, 294912, 64, 64, 0, 0, 1, 1);
    bnorm(XH2, XH2, P_(fb + 10), P_(fb + 9), P_(fb + 0), 4608, 64);
    k_fdp_merge<<<NB(T_ELEMS, 256), 256, 0, stream>>>(XL2, XH2, T0);
    bnorm(T0, outf, P_(fb + 7), P_(fb + 6), P_(fb + 5), 8, PIX);
  };
  auto sca = [&](const float* xb, float* sig) {
    k_pool_bc<<<dim3(64, 8), 256, 0, stream>>>(xb, PMEAN, PMAX);
    k_sca<<<8, 64, 0, stream>>>(PMEAN, PMAX, P_(76), P_(77), sig);
  };
  auto tca = [&](const float* xb, float* sig) {
    k_ln_ch<<<NB(8L * PIX, 256), 256, 0, stream>>>(xb, P_(80), P_(79), T0);
    gemm(T0, T0, 64, 64, QKVP, nullptr, QKV, 8L * PIX, PIX, 192, 0, 0, 0, 0);
    k_zero<<<NB(4608, 256), 256, 0, stream>>>(G, 4608);  // G + MV
    k_tca_gram<<<dim3(64, 72), 64, 0, stream>>>(QKV, G, MV);
    k_tca_finish<<<8, 64, 0, stream>>>(G, MV, P_(82), P_(81), sig);
  };
  auto ssa = [&](const float* xb, float* map) {
    k_chanpool<<<NB(8L * PIX, 256), 256, 0, stream>>>(xb, Y2);
    k_conv7<<<NB(8L * PIX, 256), 256, 0, stream>>>(Y2, P_(78), map);
  };
  auto tsa = [&](const float* xb, float* map) {
    k_ln_win<<<NB(294912L, 256), 256, 0, stream>>>(xb, P_(85), P_(84), T0);
    gemm(T0, T0, 64, 64, QWP, P_(88), QKV, 294912, PIX, 192, 1, 1, 0, 0);
    k_win_attn<<<4608, 256, 0, stream>>>(QKV, UVEC, map);
  };
  auto mulc = [&](float* xb, const float* m) { k_mul_bc<<<NB(T_ELEMS, 256), 256, 0, stream>>>(xb, m, 0); };
  auto mulp = [&](float* xb, const float* m) { k_mul_bc<<<NB(T_ELEMS, 256), 256, 0, stream>>>(xb, m, 1); };

  // ---- forward ----
  stage(X, BUFS, 0);
  fstage(X, BUFF, 0);
  sca(BUFS, SIGA); sca(BUFF, SIGB);
  mulc(BUFF, SIGA); mulc(BUFS, SIGB);

  stage(BUFS, BUFS, 1);
  fstage(BUFF, BUFF, 1);
  tca(BUFS, SIGA); tca(BUFF, SIGB);
  mulc(BUFF, SIGA); mulc(BUFS, SIGB);

  stage(BUFS, BUFS, 2);
  fstage(BUFF, BUFF, 2);
  ssa(BUFS, MAPA); ssa(BUFF, MAPB);
  mulp(BUFF, MAPA); mulp(BUFS, MAPB);

  stage(BUFS, BUFS, 3);
  fstage(BUFF, BUFF, 3);
  k_rowmean<<<1, 64, 0, stream>>>(P_(87), P_(86), UVEC);
  tsa(BUFS, MAPA); tsa(BUFF, MAPB);
  mulp(BUFF, MAPA); mulp(BUFS, MAPB);

  // fuse + final norm + residual
  gemm(BUFS, BUFF, 64, 128, FUP, P_(52), T0, 8L * PIX, PIX, 64, 0, 0, 0, 0);
  bnorm(T0, T1, P_(54), P_(53), P_(51), 8, PIX);
  {
    long cnt = 8L * PIX;
    k_zero<<<1, 256, 0, stream>>>(stats, 128);
    k_bn_stats<<<dim3(64, (unsigned)((cnt + 4095) / 4096)), 256, 0, stream>>>(T1, PIX, cnt, stats);
    k_bn_add<<<NB(T_ELEMS, 256), 256, 0, stream>>>(T1, stats, P_(50), P_(49), X,
                                                   1.f / (float)cnt, (float*)d_out);
  }
}
